// NeuralPatternClustering_429496730296
// MI455X (gfx1250) — compile-verified, matrix operand order checked by reading, not run on hardware
//
#include <hip/hip_runtime.h>

typedef __attribute__((ext_vector_type(16))) __bf16       v16bf;
typedef __attribute__((ext_vector_type(8)))  float        v8f;
typedef __attribute__((ext_vector_type(4)))  unsigned int u32x4;

#define B_ROWS   32768
#define D_DIM    1024
#define K_DIM    4096
#define BM       128
#define BN       128
#define DK       32
#define ROW_BYTES 80          // 32 bf16 payload padded to 40 bf16 -> conflict-free ds_load_b128
#define N_KT     (K_DIM / BN) // 32
#define N_DT     (D_DIM / DK) // 32

// ---- bf16 split helpers (round-to-nearest-even, inputs are finite normals) ----
static __device__ __forceinline__ unsigned short f2bf(float x) {
    unsigned int u = __builtin_bit_cast(unsigned int, x);
    unsigned int r = u + 0x7FFFu + ((u >> 16) & 1u);
    return (unsigned short)(r >> 16);
}
static __device__ __forceinline__ float bf2f(unsigned short h) {
    unsigned int u = ((unsigned int)h) << 16;
    return __builtin_bit_cast(float, u);
}
static __device__ __forceinline__ v16bf load_frag(const char* p) {
    union { u32x4 u[2]; v16bf v; } f;
    f.u[0] = *(const u32x4*)(p);
    f.u[1] = *(const u32x4*)(p + 32);
    return f.v;
}

// ---- prepass: row-wise sum of squares (used for centroids: csq[k] = ||c_k||^2) ----
__global__ void rowsq_kernel(const float* __restrict__ mat, float* __restrict__ out, int cols) {
    __shared__ float red[128];
    const int row = blockIdx.x;
    const float* p = mat + (size_t)row * cols;
    float s = 0.f;
    for (int c = threadIdx.x; c < cols; c += 128) { float x = p[c]; s += x * x; }
    red[threadIdx.x] = s;
    __syncthreads();
    for (int st = 64; st > 0; st >>= 1) {
        if (threadIdx.x < st) red[threadIdx.x] += red[threadIdx.x + st];
        __syncthreads();
    }
    if (threadIdx.x == 0) out[row] = red[0];
}

// ---- main fused kernel: bf16x3 GEMM + distance + streaming softmax/min ----
__global__ __launch_bounds__(256) void cluster_main(
    const float* __restrict__ features,
    const float* __restrict__ centroids,
    const float* __restrict__ W,
    const float* __restrict__ bias,
    const float* __restrict__ csq,
    float* __restrict__ logits,
    float* __restrict__ pen_partials)
{
    __shared__ __align__(16) char sAhi[BM * ROW_BYTES];
    __shared__ __align__(16) char sAlo[BM * ROW_BYTES];
    __shared__ __align__(16) char sBhi[BN * ROW_BYTES];
    __shared__ __align__(16) char sBlo[BN * ROW_BYTES];
    __shared__ float sFsqPart[256];
    __shared__ float sFsq[BM];
    __shared__ float sRedSum[BM * 2];
    __shared__ float sRedW[BM * 2];
    __shared__ float sRedMin[BM * 2];
    __shared__ float sPen[BM];

    const int t       = threadIdx.x;
    const int rowBase = blockIdx.x * BM;
    const int lane    = t & 31;
    const int wave    = t >> 5;
    const int wm      = wave & 3;   // M wave: rows wm*32 .. +31
    const int wn      = wave >> 2;  // N wave: cols wn*64 .. +63
    const int half    = lane >> 4;  // K-half selector (A/B frag layout)
    const int r16     = lane & 15;

    // staging assignment: thread -> (row, 16-col half)
    const int srow = t >> 1;
    const int scol = (t & 1) * 16;

    float state_min[2][8], state_sum[2][8], state_w[2][8], fsq_reg[2][8];
#pragma unroll
    for (int mt = 0; mt < 2; ++mt)
#pragma unroll
        for (int r = 0; r < 8; ++r) {
            state_min[mt][r] = 3.0e38f;
            state_sum[mt][r] = 0.f;
            state_w[mt][r]   = 0.f;
            fsq_reg[mt][r]   = 0.f;
        }

    for (int kt = 0; kt < N_KT; ++kt) {
        v8f acc[2][4];
#pragma unroll
        for (int mt = 0; mt < 2; ++mt)
#pragma unroll
            for (int nt = 0; nt < 4; ++nt)
#pragma unroll
                for (int r = 0; r < 8; ++r) acc[mt][nt][r] = 0.f;

        float fsqp = 0.f;

        for (int dt = 0; dt < N_DT; ++dt) {
            const int dk = dt * DK;
            __syncthreads();  // previous compute done before overwriting tiles

            // ---- stage A (features) : f32 -> bf16 hi/lo ----
            {
                const float* gp = features + (size_t)(rowBase + srow) * D_DIM + dk + scol;
                float x[16];
                *(float4*)(x + 0)  = ((const float4*)gp)[0];
                *(float4*)(x + 4)  = ((const float4*)gp)[1];
                *(float4*)(x + 8)  = ((const float4*)gp)[2];
                *(float4*)(x + 12) = ((const float4*)gp)[3];
                if (kt == 0) {
#pragma unroll
                    for (int j = 0; j < 16; ++j) fsqp += x[j] * x[j];
                }
                u32x4 vh0, vh1, vl0, vl1;
#pragma unroll
                for (int j = 0; j < 4; ++j) {
                    unsigned short h0 = f2bf(x[2*j]),     h1 = f2bf(x[2*j+1]);
                    unsigned short l0 = f2bf(x[2*j]   - bf2f(h0));
                    unsigned short l1 = f2bf(x[2*j+1] - bf2f(h1));
                    vh0[j] = (unsigned)h0 | ((unsigned)h1 << 16);
                    vl0[j] = (unsigned)l0 | ((unsigned)l1 << 16);
                }
#pragma unroll
                for (int j = 0; j < 4; ++j) {
                    unsigned short h0 = f2bf(x[8+2*j]),     h1 = f2bf(x[8+2*j+1]);
                    unsigned short l0 = f2bf(x[8+2*j]   - bf2f(h0));
                    unsigned short l1 = f2bf(x[8+2*j+1] - bf2f(h1));
                    vh1[j] = (unsigned)h0 | ((unsigned)h1 << 16);
                    vl1[j] = (unsigned)l0 | ((unsigned)l1 << 16);
                }
                char* dh = sAhi + srow * ROW_BYTES + scol * 2;
                char* dl = sAlo + srow * ROW_BYTES + scol * 2;
                *(u32x4*)(dh) = vh0; *(u32x4*)(dh + 16) = vh1;
                *(u32x4*)(dl) = vl0; *(u32x4*)(dl + 16) = vl1;
            }
            // ---- stage B (centroids) ----
            {
                const float* gp = centroids + (size_t)(kt * BN + srow) * D_DIM + dk + scol;
                float x[16];
                *(float4*)(x + 0)  = ((const float4*)gp)[0];
                *(float4*)(x + 4)  = ((const float4*)gp)[1];
                *(float4*)(x + 8)  = ((const float4*)gp)[2];
                *(float4*)(x + 12) = ((const float4*)gp)[3];
                u32x4 vh0, vh1, vl0, vl1;
#pragma unroll
                for (int j = 0; j < 4; ++j) {
                    unsigned short h0 = f2bf(x[2*j]),     h1 = f2bf(x[2*j+1]);
                    unsigned short l0 = f2bf(x[2*j]   - bf2f(h0));
                    unsigned short l1 = f2bf(x[2*j+1] - bf2f(h1));
                    vh0[j] = (unsigned)h0 | ((unsigned)h1 << 16);
                    vl0[j] = (unsigned)l0 | ((unsigned)l1 << 16);
                }
#pragma unroll
                for (int j = 0; j < 4; ++j) {
                    unsigned short h0 = f2bf(x[8+2*j]),     h1 = f2bf(x[8+2*j+1]);
                    unsigned short l0 = f2bf(x[8+2*j]   - bf2f(h0));
                    unsigned short l1 = f2bf(x[8+2*j+1] - bf2f(h1));
                    vh1[j] = (unsigned)h0 | ((unsigned)h1 << 16);
                    vl1[j] = (unsigned)l0 | ((unsigned)l1 << 16);
                }
                char* dh = sBhi + srow * ROW_BYTES + scol * 2;
                char* dl = sBlo + srow * ROW_BYTES + scol * 2;
                *(u32x4*)(dh) = vh0; *(u32x4*)(dh + 16) = vh1;
                *(u32x4*)(dl) = vl0; *(u32x4*)(dl + 16) = vl1;
            }
            __syncthreads();

            // ---- bf16x3 WMMA: acc += Ahi*Bhi + Ahi*Blo + Alo*Bhi ----
            v16bf ahi[2], alo[2];
#pragma unroll
            for (int mt = 0; mt < 2; ++mt) {
                const int arow = wm * 32 + mt * 16 + r16;
                ahi[mt] = load_frag(sAhi + arow * ROW_BYTES + half * 16);
                alo[mt] = load_frag(sAlo + arow * ROW_BYTES + half * 16);
            }
#pragma unroll
            for (int nt = 0; nt < 4; ++nt) {
                const int brow = wn * 64 + nt * 16 + r16;
                v16bf bhi = load_frag(sBhi + brow * ROW_BYTES + half * 16);
                v16bf blo = load_frag(sBlo + brow * ROW_BYTES + half * 16);
#pragma unroll
                for (int mt = 0; mt < 2; ++mt) {
                    acc[mt][nt] = __builtin_amdgcn_wmma_f32_16x16x32_bf16(
                        false, ahi[mt], false, bhi, (short)0, acc[mt][nt], false, false);
                    acc[mt][nt] = __builtin_amdgcn_wmma_f32_16x16x32_bf16(
                        false, ahi[mt], false, blo, (short)0, acc[mt][nt], false, false);
                    acc[mt][nt] = __builtin_amdgcn_wmma_f32_16x16x32_bf16(
                        false, alo[mt], false, bhi, (short)0, acc[mt][nt], false, false);
                }
            }
        } // D loop

        if (kt == 0) {  // finish ||f||^2 for this row block (computed during staging)
            __syncthreads();
            sFsqPart[t] = fsqp;
            __syncthreads();
            if (t < BM) sFsq[t] = sFsqPart[2 * t] + sFsqPart[2 * t + 1];
            __syncthreads();
#pragma unroll
            for (int mt = 0; mt < 2; ++mt)
#pragma unroll
                for (int r = 0; r < 8; ++r)
                    fsq_reg[mt][r] = sFsq[wm * 32 + mt * 16 + half * 8 + r];
        }

        // ---- epilogue: dist -> running min / sum(exp) / sum(exp*W) ----
#pragma unroll
        for (int nt = 0; nt < 4; ++nt) {
            const int kg = kt * BN + wn * 64 + nt * 16 + r16;  // C layout: N = lane&15
            const float cs = csq[kg];
            const float wv = W[kg];
#pragma unroll
            for (int mt = 0; mt < 2; ++mt)
#pragma unroll
                for (int r = 0; r < 8; ++r) {
                    float d2 = fmaxf(fsq_reg[mt][r] + cs - 2.0f * acc[mt][nt][r], 0.0f);
                    float d  = sqrtf(d2);
                    state_min[mt][r] = fminf(state_min[mt][r], d);
                    float e = __expf(-d);                 // TEMPERATURE = 1.0
                    state_sum[mt][r] += e;
                    state_w[mt][r] = __builtin_fmaf(e, wv, state_w[mt][r]);
                }
        }
    } // K-tile loop

    // ---- row reductions: across the 16 lanes of each C half, then across N-waves ----
#pragma unroll
    for (int mt = 0; mt < 2; ++mt)
#pragma unroll
        for (int r = 0; r < 8; ++r) {
            float s = state_sum[mt][r], w = state_w[mt][r], m = state_min[mt][r];
            for (int off = 1; off < 16; off <<= 1) {
                s += __shfl_xor(s, off, 32);
                w += __shfl_xor(w, off, 32);
                m = fminf(m, __shfl_xor(m, off, 32));
            }
            if (r16 == 0) {
                const int mloc = wm * 32 + mt * 16 + half * 8 + r;
                sRedSum[mloc * 2 + wn] = s;
                sRedW[mloc * 2 + wn]   = w;
                sRedMin[mloc * 2 + wn] = m;
            }
        }
    __syncthreads();

    if (t < BM) {
        const float s = sRedSum[2 * t] + sRedSum[2 * t + 1];
        const float w = sRedW[2 * t] + sRedW[2 * t + 1];
        const float m = fminf(sRedMin[2 * t], sRedMin[2 * t + 1]);
        logits[rowBase + t] = w / s + bias[0];
        sPen[t] = m;
    }
    __syncthreads();
    for (int st = BM / 2; st > 0; st >>= 1) {
        if (t < st) sPen[t] += sPen[t + st];
        __syncthreads();
    }
    if (t == 0) pen_partials[blockIdx.x] = sPen[0];
}

// ---- deterministic penalty reduction ----
__global__ void finalize_kernel(const float* __restrict__ partials, float* __restrict__ pen_out, int n) {
    __shared__ float red[256];
    float s = 0.f;
    for (int i = threadIdx.x; i < n; i += 256) s += partials[i];
    red[threadIdx.x] = s;
    __syncthreads();
    for (int st = 128; st > 0; st >>= 1) {
        if (threadIdx.x < st) red[threadIdx.x] += red[threadIdx.x + st];
        __syncthreads();
    }
    if (threadIdx.x == 0) pen_out[0] = red[0] / (float)B_ROWS;
}

extern "C" void kernel_launch(void* const* d_in, const int* in_sizes, int n_in,
                              void* d_out, int out_size, void* d_ws, size_t ws_size,
                              hipStream_t stream) {
    (void)in_sizes; (void)n_in; (void)out_size; (void)ws_size;
    const float* features  = (const float*)d_in[0];  // [32768,1024]
    const float* centroids = (const float*)d_in[1];  // [4096,1024]
    const float* W         = (const float*)d_in[2];  // [1,4096]
    const float* b         = (const float*)d_in[3];  // [1]
    float* out      = (float*)d_out;                 // [32768 logits][1 penalty]
    float* csq      = (float*)d_ws;                  // 4096 floats
    float* partials = csq + K_DIM;                   // 256 floats

    rowsq_kernel<<<K_DIM, 128, 0, stream>>>(centroids, csq, D_DIM);
    cluster_main<<<B_ROWS / BM, 256, 0, stream>>>(features, centroids, W, b, csq, out, partials);
    finalize_kernel<<<1, 256, 0, stream>>>(partials, out + B_ROWS, B_ROWS / BM);
}
